// LowRankAttention_14181982011820
// MI455X (gfx1250) — matrix operand order repaired and checked by reading, not compiled, not run on hardware
//
#include <hip/hip_runtime.h>

// ---------------------------------------------------------------------------
// LowRankAttention fused implementation for MI455X (gfx1250, wave32, WMMA).
// All GEMMs use v_wmma_f32_16x16x32_bf16 (bf16 in, f32 accumulate).
// ---------------------------------------------------------------------------

#define HNUM 8
#define DKD  64
#define DVD  64
#define MIDD 128
#define DMD  512
#define BSZ  8
#define NQ   100
#define NK   100
#define NROWS (BSZ * NQ)   // 800

typedef __attribute__((ext_vector_type(16))) __bf16       v16bf;
typedef __attribute__((ext_vector_type(8)))  float        v8f;
typedef __attribute__((ext_vector_type(4)))  float        f32x4;
typedef __attribute__((ext_vector_type(4)))  unsigned int u32x4;

union AFrag { u32x4 u[2]; v16bf v; };

__device__ __forceinline__ f32x4 ld4(const float* p) { return *(const f32x4*)p; }

__device__ __forceinline__ void cvt8(v16bf& d, int base, f32x4 a, f32x4 b) {
  d[base + 0] = (__bf16)a.x; d[base + 1] = (__bf16)a.y;
  d[base + 2] = (__bf16)a.z; d[base + 3] = (__bf16)a.w;
  d[base + 4] = (__bf16)b.x; d[base + 5] = (__bf16)b.y;
  d[base + 6] = (__bf16)b.z; d[base + 7] = (__bf16)b.w;
}

__device__ __forceinline__ void cvt8mul(v16bf& d, int base,
                                        f32x4 ka, f32x4 kb, f32x4 qa, f32x4 qb) {
  d[base + 0] = (__bf16)(ka.x * qa.x); d[base + 1] = (__bf16)(ka.y * qa.y);
  d[base + 2] = (__bf16)(ka.z * qa.z); d[base + 3] = (__bf16)(ka.w * qa.w);
  d[base + 4] = (__bf16)(kb.x * qb.x); d[base + 5] = (__bf16)(kb.y * qb.y);
  d[base + 6] = (__bf16)(kb.z * qb.z); d[base + 7] = (__bf16)(kb.w * qb.w);
}

__device__ __forceinline__ v8f wmma_bf16(v16bf a, v16bf b, v8f c) {
  // D = A(16x32 bf16) x B(32x16 bf16) + C(16x16 f32)
  return __builtin_amdgcn_wmma_f32_16x16x32_bf16(false, a, false, b,
                                                 (short)0, c, false, false);
}

// ---------------------------------------------------------------------------
// Kernel 1: in-projection  y = GroupNormAffine(ELU(x @ W + b))
// Block: 16 rows x 64 cols (one head group) -> GroupNorm is block-local.
// Grid: (50 row tiles, 8 groups). 128 threads = 4 waves, one 16x16 N-tile each.
// Output layout: dst[((b*H + grp)*NQ + n)*64 + d]
// ---------------------------------------------------------------------------
__global__ __launch_bounds__(128) void proj_kernel(
    const float* __restrict__ X, const float* __restrict__ W,
    const float* __restrict__ bias, const float* __restrict__ gw,
    const float* __restrict__ gb, float* __restrict__ dst) {
  __shared__ __attribute__((aligned(16))) __bf16 xs[16][DMD];  // 16 KB
  __shared__ __attribute__((aligned(16))) float  ys[16][64];   // 4 KB
  __shared__ float mu_s[16], rv_s[16];

  const int tid  = threadIdx.x;
  const int r0   = blockIdx.x * 16;   // global row base (800 = 50*16, exact)
  const int grp  = blockIdx.y;        // head group 0..7

  // Stage x tile (16 x 512 fp32 -> bf16 in LDS, A-matrix friendly row-major)
  for (int i = tid; i < 16 * (DMD / 4); i += 128) {
    int row = i >> 7;              // 128 f32x4 per row
    int c4  = (i & 127) << 2;
    f32x4 f = ld4(X + (r0 + row) * DMD + c4);
    xs[row][c4 + 0] = (__bf16)f.x; xs[row][c4 + 1] = (__bf16)f.y;
    xs[row][c4 + 2] = (__bf16)f.z; xs[row][c4 + 3] = (__bf16)f.w;
  }
  __syncthreads();

  const int wave = tid >> 5, lane = tid & 31;
  const int n0   = wave * 16;                 // col offset within group
  const int c0   = grp * 64 + n0;             // global col base of this tile
  const int arow = lane & 15;                 // A-layout: M = lane&15
  const int hi   = lane >> 4;
  const int off  = hi * 8;                    // A-layout K offset

  v8f c = {0.f, 0.f, 0.f, 0.f, 0.f, 0.f, 0.f, 0.f};
  for (int k0 = 0; k0 < DMD; k0 += 32) {
    AFrag a;
    a.u[0] = *(const u32x4*)&xs[arow][k0 + off];       // K = k0+off .. +7
    a.u[1] = *(const u32x4*)&xs[arow][k0 + 16 + off];  // K = k0+16+off .. +7
    // B: lane holds column N = lane&15, K = k0+off..+7 and k0+16+off..+7
    const float* wp = W + (k0 + off) * DMD + c0 + (lane & 15);
    v16bf bf;
    for (int s = 0; s < 8; s++) { bf[s] = (__bf16)wp[s * DMD]; bf[8 + s] = (__bf16)wp[(16 + s) * DMD]; }
    c = wmma_bf16(a.v, bf, c);
  }

  // bias + ELU into LDS (C/D layout: row = r + hi*8, col = lane&15)
  const int colg = n0 + (lane & 15);          // col within group, 0..63
  const float bv = bias[grp * 64 + colg];
  #pragma unroll
  for (int r = 0; r < 8; r++) {
    int row = r + hi * 8;
    float y = c[r] + bv;
    y = (y > 0.f) ? y : (expf(y) - 1.f);      // ELU(alpha=1)
    ys[row][colg] = y;
  }
  __syncthreads();

  if (tid < 16) {                             // per-row GroupNorm stats (64 ch)
    float s = 0.f, s2 = 0.f;
    for (int j = 0; j < 64; j++) { float v = ys[tid][j]; s += v; s2 += v * v; }
    float m = s * (1.f / 64.f);
    mu_s[tid] = m;
    rv_s[tid] = rsqrtf(fmaxf(s2 * (1.f / 64.f) - m * m, 0.f) + 1e-5f);
  }
  __syncthreads();

  for (int i = tid; i < 16 * 64; i += 128) {
    int row = i >> 6, col = i & 63;
    int gc  = grp * 64 + col;
    float v = (ys[row][col] - mu_s[row]) * rv_s[row] * gw[gc] + gb[gc];
    int rg = r0 + row;
    int b = rg / NQ, n = rg % NQ;
    dst[((b * HNUM + grp) * NQ + n) * 64 + col] = v;
  }
}

// ---------------------------------------------------------------------------
// Kernel 2: fused attention core. One block per (b, h, 16-q tile).
// 256 threads = 8 waves; wave w owns MID columns [16w, 16w+16).
// W_m B-fragments live in registers for the whole kernel.
// attn_map tiles exist only in WMMA accumulators (never hits memory).
// ---------------------------------------------------------------------------
__global__ __launch_bounds__(256) void attn_kernel(
    const float* __restrict__ q1p, const float* __restrict__ q2p,
    const float* __restrict__ kp,  const float* __restrict__ vp,
    const float* __restrict__ Wm,  const float* __restrict__ bm,
    const float* __restrict__ Ws,  const float* __restrict__ Wc,
    const float* __restrict__ bc,  float* __restrict__ attn_buf) {
  __shared__ __attribute__((aligned(16))) float Ks[112][DKD];   // 28 KB (zero-padded)
  __shared__ __attribute__((aligned(16))) float Vs[112][DVD];   // 28 KB
  __shared__ __attribute__((aligned(16))) float q1s[16][DKD];   // 4 KB
  __shared__ __attribute__((aligned(16))) float q2s[16][DVD];   // 4 KB
  __shared__ float s_acc[112];   // spatial logits / softmax probs
  __shared__ float pool[MIDD];   // channel pool accumulator

  const int tid = threadIdx.x;
  const int qt  = blockIdx.x;          // 0..6
  const int bh  = blockIdx.y;          // 0..63
  const int b   = bh >> 3, h = bh & 7;
  const int q0  = qt * 16;

  const float* Kg  = kp  + bh * NK * DKD;
  const float* Vg  = vp  + bh * NK * DVD;
  const float* Q1g = q1p + bh * NQ * DKD;
  const float* Q2g = q2p + bh * NQ * DVD;

  const f32x4 z4 = {0.f, 0.f, 0.f, 0.f};
  for (int i = tid; i < 112 * (DKD / 4); i += 256) {   // stage K,V (pad k>=100 with 0)
    int row = i >> 4, c4 = (i & 15) << 2;
    f32x4 kv = (row < NK) ? ld4(Kg + row * DKD + c4) : z4;
    f32x4 vv = (row < NK) ? ld4(Vg + row * DVD + c4) : z4;
    *(f32x4*)&Ks[row][c4] = kv;
    *(f32x4*)&Vs[row][c4] = vv;
  }
  for (int i = tid; i < 16 * (DKD / 4); i += 256) {    // stage q1,q2 tile
    int row = i >> 4, c4 = (i & 15) << 2;
    int q = q0 + row;
    f32x4 q1v = (q < NQ) ? ld4(Q1g + q * DKD + c4) : z4;
    f32x4 q2v = (q < NQ) ? ld4(Q2g + q * DVD + c4) : z4;
    *(f32x4*)&q1s[row][c4] = q1v;
    *(f32x4*)&q2s[row][c4] = q2v;
  }
  __syncthreads();

  const int wave = tid >> 5, lane = tid & 31;
  const int m0   = wave * 16;          // MID chunk base
  const int mcol = lane & 15;
  const int hi   = lane >> 4;
  const int off  = hi * 8;

  // Preload W_m (64 x 128) B-fragments: d-steps [0,32) and [32,64)
  v16bf bfr0, bfr1;
  {
    const float* wp0 = Wm + off * MIDD + m0 + mcol;    // m = m0+mcol, d = off+s / 16+off+s
    for (int s = 0; s < 8; s++) {
      bfr0[s] = (__bf16)wp0[s * MIDD]; bfr0[8 + s] = (__bf16)wp0[(16 + s) * MIDD];
      bfr1[s] = (__bf16)wp0[(32 + s) * MIDD]; bfr1[8 + s] = (__bf16)wp0[(48 + s) * MIDD];
    }
  }
  const float bmv = bm[m0 + mcol];
  const float wsv = Ws[m0 + mcol];

  for (int qq = 0; qq < 16; qq++) {
    const int q = q0 + qq;
    if (q >= NQ) break;                 // uniform across block

    if (tid < MIDD) pool[tid] = 0.f;
    if (tid < 112)  s_acc[tid] = 0.f;
    __syncthreads();

    for (int kt = 0; kt < 7; kt++) {
      const int k0 = kt * 16;
      const int krow = k0 + mcol;       // A-layout: M = lane&15 -> key index

      v16bf a0, a1;                     // A = K[k,:] * q1[q,:]  (on the fly)
      cvt8mul(a0, 0, ld4(&Ks[krow][off]),      ld4(&Ks[krow][off + 4]),
                     ld4(&q1s[qq][off]),       ld4(&q1s[qq][off + 4]));
      cvt8mul(a0, 8, ld4(&Ks[krow][16 + off]), ld4(&Ks[krow][20 + off]),
                     ld4(&q1s[qq][16 + off]),  ld4(&q1s[qq][20 + off]));
      cvt8mul(a1, 0, ld4(&Ks[krow][32 + off]), ld4(&Ks[krow][36 + off]),
                     ld4(&q1s[qq][32 + off]),  ld4(&q1s[qq][36 + off]));
      cvt8mul(a1, 8, ld4(&Ks[krow][48 + off]), ld4(&Ks[krow][52 + off]),
                     ld4(&q1s[qq][48 + off]),  ld4(&q1s[qq][52 + off]));

      v8f c = {0.f, 0.f, 0.f, 0.f, 0.f, 0.f, 0.f, 0.f};
      c = wmma_bf16(a0, bfr0, c);
      c = wmma_bf16(a1, bfr1, c);

      // relu(map + b_m); fold into s[k] and pool[m]
      float pv = 0.f;
      #pragma unroll
      for (int r = 0; r < 8; r++) {
        int kk = k0 + r + hi * 8;
        float vmap = c[r] + bmv;
        vmap = (vmap > 0.f) ? vmap : 0.f;
        if (kk >= NK) vmap = 0.f;       // mask padded keys
        pv += vmap;
        float sv = vmap * wsv;          // reduce over 16 m-lanes
        sv += __shfl_xor(sv, 1); sv += __shfl_xor(sv, 2);
        sv += __shfl_xor(sv, 4); sv += __shfl_xor(sv, 8);
        if (mcol == 0) atomicAdd(&s_acc[kk], sv);
      }
      pv += __shfl_xor(pv, 16);         // combine the two k-row halves (same m)
      if (hi == 0) atomicAdd(&pool[m0 + mcol], pv);
    }
    __syncthreads();

    if (tid == 0) {                     // softmax over k (b_s is softmax-invariant)
      float mx = -1e30f;
      for (int k = 0; k < NK; k++) mx = fmaxf(mx, s_acc[k]);
      float Z = 0.f;
      for (int k = 0; k < NK; k++) { float e = expf(s_acc[k] - mx); s_acc[k] = e; Z += e; }
      float inv = 1.f / Z;
      for (int k = 0; k < NK; k++) s_acc[k] *= inv;
    }
    __syncthreads();

    if (tid < DVD) {                    // p@V, channel gate, q2 gate, store
      const int dv = tid;
      float acc = 0.f;
      for (int k = 0; k < NK; k++) acc += s_acc[k] * Vs[k][dv];
      float cg = bc[dv];
      for (int m = 0; m < MIDD; m++) cg += (pool[m] * 0.01f) * Wc[m * DVD + dv];
      float gate = 1.f / (1.f + expf(-cg));
      attn_buf[(b * NQ + q) * (HNUM * DVD) + h * DVD + dv] =
          acc * q2s[qq][dv] * gate;
    }
    __syncthreads();
  }
}

// ---------------------------------------------------------------------------
// Kernel 3: output projection  out = attn_buf(800x512) @ W_o + b_o
// ---------------------------------------------------------------------------
__global__ __launch_bounds__(128) void outproj_kernel(
    const float* __restrict__ X, const float* __restrict__ W,
    const float* __restrict__ bias, float* __restrict__ out) {
  __shared__ __attribute__((aligned(16))) __bf16 xs[16][DMD];

  const int tid = threadIdx.x;
  const int r0  = blockIdx.x * 16;
  const int cg0 = blockIdx.y * 64;

  for (int i = tid; i < 16 * (DMD / 4); i += 128) {
    int row = i >> 7, c4 = (i & 127) << 2;
    f32x4 f = ld4(X + (r0 + row) * DMD + c4);
    xs[row][c4 + 0] = (__bf16)f.x; xs[row][c4 + 1] = (__bf16)f.y;
    xs[row][c4 + 2] = (__bf16)f.z; xs[row][c4 + 3] = (__bf16)f.w;
  }
  __syncthreads();

  const int wave = tid >> 5, lane = tid & 31;
  const int c0   = cg0 + wave * 16;
  const int arow = lane & 15;
  const int hi   = lane >> 4;
  const int off  = hi * 8;

  v8f c = {0.f, 0.f, 0.f, 0.f, 0.f, 0.f, 0.f, 0.f};
  for (int k0 = 0; k0 < DMD; k0 += 32) {
    AFrag a;
    a.u[0] = *(const u32x4*)&xs[arow][k0 + off];
    a.u[1] = *(const u32x4*)&xs[arow][k0 + 16 + off];
    const float* wp = W + (k0 + off) * DMD + c0 + (lane & 15);
    v16bf bf;
    for (int s = 0; s < 8; s++) { bf[s] = (__bf16)wp[s * DMD]; bf[8 + s] = (__bf16)wp[(16 + s) * DMD]; }
    c = wmma_bf16(a.v, bf, c);
  }

  const int colg = c0 + (lane & 15);
  const float bv = bias[colg];
  #pragma unroll
  for (int r = 0; r < 8; r++) {
    int row = r + hi * 8;
    out[(r0 + row) * DMD + colg] = c[r] + bv;
  }
}

// ---------------------------------------------------------------------------
extern "C" void kernel_launch(void* const* d_in, const int* in_sizes, int n_in,
                              void* d_out, int out_size, void* d_ws, size_t ws_size,
                              hipStream_t stream) {
  const float* queries = (const float*)d_in[0];
  const float* keys    = (const float*)d_in[1];
  const float* values  = (const float*)d_in[2];
  const float* W_q1 = (const float*)d_in[3];
  const float* b_q1 = (const float*)d_in[4];
  const float* g_q1w = (const float*)d_in[5];
  const float* g_q1b = (const float*)d_in[6];
  const float* W_q2 = (const float*)d_in[7];
  const float* b_q2 = (const float*)d_in[8];
  const float* g_q2w = (const float*)d_in[9];
  const float* g_q2b = (const float*)d_in[10];
  const float* W_k  = (const float*)d_in[11];
  const float* b_k  = (const float*)d_in[12];
  const float* g_kw = (const float*)d_in[13];
  const float* g_kb = (const float*)d_in[14];
  const float* W_v  = (const float*)d_in[15];
  const float* b_v  = (const float*)d_in[16];
  const float* g_vw = (const float*)d_in[17];
  const float* g_vb = (const float*)d_in[18];
  const float* W_m  = (const float*)d_in[19];
  const float* b_m  = (const float*)d_in[20];
  const float* W_s  = (const float*)d_in[21];
  // d_in[22] = b_s : softmax-invariant constant, intentionally unused
  const float* W_c  = (const float*)d_in[23];
  const float* b_c  = (const float*)d_in[24];
  const float* W_o  = (const float*)d_in[25];
  const float* b_o  = (const float*)d_in[26];

  // Workspace: 5 fp32 buffers of 8*8*100*64 = 409600 elems (8 MB total)
  const size_t NBH = (size_t)BSZ * HNUM * NQ * 64;
  float* q1p = (float*)d_ws;
  float* q2p = q1p + NBH;
  float* kp  = q2p + NBH;
  float* vp  = kp + NBH;
  float* attn_buf = vp + NBH;   // (800 x 512)

  dim3 pgrid(NROWS / 16, HNUM);
  proj_kernel<<<pgrid, 128, 0, stream>>>(queries, W_q1, b_q1, g_q1w, g_q1b, q1p);
  proj_kernel<<<pgrid, 128, 0, stream>>>(queries, W_q2, b_q2, g_q2w, g_q2b, q2p);
  proj_kernel<<<pgrid, 128, 0, stream>>>(keys,    W_k,  b_k,  g_kw,  g_kb,  kp);
  proj_kernel<<<pgrid, 128, 0, stream>>>(values,  W_v,  b_v,  g_vw,  g_vb,  vp);

  attn_kernel<<<dim3(7, BSZ * HNUM), 256, 0, stream>>>(
      q1p, q2p, kp, vp, W_m, b_m, W_s, W_c, b_c, attn_buf);

  outproj_kernel<<<dim3(NROWS / 16, DMD / 64), 128, 0, stream>>>(
      attn_buf, W_o, b_o, (float*)d_out);
}
